// GraphSAGE_10514079941583
// MI455X (gfx1250) — matrix operand order repaired and checked by reading, not compiled, not run on hardware
//
#include <hip/hip_runtime.h>

// ---------------------------------------------------------------------------
// GraphSAGE 2-layer forward for MI455X (gfx1250, wave32).
//   layer: out = mean_agg(x) @ W_l + x @ W_r + b   (+ ReLU after layer 1)
//
// Roofline: GEMMs are only ~13 GFLOP; traffic dominates. Working set ~103MB
// fits the 192MB L2, so edge gather + global_atomic_add_f32 scatter run at
// L2 bandwidth. GEMM operands are pre-converted to bf16 (weights transposed
// to [N,K]) so WMMA fragments load as contiguous b128s; the K-loop is
// 8x v_wmma_f32_16x16x32_bf16 per step with no in-loop conversion VALU.
// ---------------------------------------------------------------------------

#define N_NODES 50000
#define N_EDGES 600000
#define IN_CH   128
#define HID_CH  256
#define OUT_CH  128

typedef __attribute__((ext_vector_type(16))) __bf16 v16bf;
typedef __attribute__((ext_vector_type(8)))  __bf16 v8bf;
typedef __attribute__((ext_vector_type(4)))  __bf16 v4bf;
typedef __attribute__((ext_vector_type(8)))  float  v8f;

__device__ __forceinline__ __bf16 cvt_bf16(float f) { return (__bf16)f; }

// ---------------------------------------------------------------------------
__global__ void zero_f32(float* __restrict__ p, long long n) {
  long long i = (long long)blockIdx.x * blockDim.x + threadIdx.x;
  long long stride = (long long)gridDim.x * blockDim.x;
  for (; i < n; i += stride) p[i] = 0.0f;
}

// cnt[dst] += 1 per edge
__global__ void degree_kernel(const int* __restrict__ dst, float* __restrict__ cnt,
                              int nEdges) {
  int e = blockIdx.x * blockDim.x + threadIdx.x;
  if (e < nEdges) atomicAdd(&cnt[dst[e]], 1.0f);
}

// agg[dst] += x[src] per edge, f32 source (layer 1)
__global__ void scatter_add_f32(const float* __restrict__ x, const int* __restrict__ src,
                                const int* __restrict__ dst, float* __restrict__ agg,
                                int nEdges, int ch) {
  const int vecPerRow = ch >> 2;
  long long tid = (long long)blockIdx.x * blockDim.x + threadIdx.x;
  if (tid >= (long long)nEdges * vecPerRow) return;
  int e = (int)(tid / vecPerRow);
  int c = (int)(tid % vecPerRow) << 2;
  int s = src[e], d = dst[e];
  float4 v = *(const float4*)(x + (long long)s * ch + c);
  float* o = agg + (long long)d * ch + c;
  atomicAdd(o + 0, v.x);
  atomicAdd(o + 1, v.y);
  atomicAdd(o + 2, v.z);
  atomicAdd(o + 3, v.w);
}

// agg[dst] += h[src] per edge, bf16 source (layer 2; halves gather bytes)
__global__ void scatter_add_bf16(const __bf16* __restrict__ h, const int* __restrict__ src,
                                 const int* __restrict__ dst, float* __restrict__ agg,
                                 int nEdges, int ch) {
  const int vecPerRow = ch >> 2;
  long long tid = (long long)blockIdx.x * blockDim.x + threadIdx.x;
  if (tid >= (long long)nEdges * vecPerRow) return;
  int e = (int)(tid / vecPerRow);
  int c = (int)(tid % vecPerRow) << 2;
  int s = src[e], d = dst[e];
  v4bf v = *(const v4bf*)(h + (long long)s * ch + c);
  float* o = agg + (long long)d * ch + c;
  atomicAdd(o + 0, (float)v[0]);
  atomicAdd(o + 1, (float)v[1]);
  atomicAdd(o + 2, (float)v[2]);
  atomicAdd(o + 3, (float)v[3]);
}

// f32 rows -> bf16 rows; optional per-row mean division (cnt != null)
__global__ void rows_to_bf16(const float* __restrict__ in, const float* __restrict__ cnt,
                             __bf16* __restrict__ out, long long nElem, int ch) {
  long long t = (long long)blockIdx.x * blockDim.x + threadIdx.x;
  long long e0 = t << 2;
  if (e0 >= nElem) return;
  float s = 1.0f;
  if (cnt) {
    int row = (int)(e0 / ch);
    s = 1.0f / fmaxf(cnt[row], 1.0f);
  }
  float4 v = *(const float4*)(in + e0);
  v4bf o;
  o[0] = cvt_bf16(v.x * s);
  o[1] = cvt_bf16(v.y * s);
  o[2] = cvt_bf16(v.z * s);
  o[3] = cvt_bf16(v.w * s);
  *(v4bf*)(out + e0) = o;
}

// W [K,N] f32 -> Wt [N,K] bf16 (fragment-friendly: lane n reads contiguous K)
__global__ void w_to_bf16t(const float* __restrict__ W, __bf16* __restrict__ Wt,
                           int K, int N) {
  int t = blockIdx.x * blockDim.x + threadIdx.x;
  if (t >= K * N) return;
  int k = t / N, n = t - k * N;
  Wt[(long long)n * K + k] = cvt_bf16(W[t]);
}

// ---------------------------------------------------------------------------
// Fused dual-GEMM: out[M,N] = Ab @ Wl^T' + Xb @ Wr^T' + bias [+relu]
//   Ab,Xb : bf16 [M,K] row-major (Ab already mean-normalized)
//   Wlt,Wrt: bf16 [N,K] (pre-transposed)
// One wave32 per (16-row M-tile x 64-col N-group): 4 accumulators, A
// fragments reused 4x, 8 WMMAs per K-step. M,N,K divisibility guaranteed.
// ---------------------------------------------------------------------------
__global__ void sage_gemm_bf16(const __bf16* __restrict__ Ab, const __bf16* __restrict__ Xb,
                               const __bf16* __restrict__ Wlt, const __bf16* __restrict__ Wrt,
                               const float* __restrict__ bias,
                               float* __restrict__ outF, __bf16* __restrict__ outB,
                               int M, int K, int N, int relu) {
  const int lane = threadIdx.x & 31;
  const int wave = threadIdx.x >> 5;
  const int nGroups = N >> 6;                       // 64 cols per wave
  const int totalJobs = (M >> 4) * nGroups;
  const int job = blockIdx.x * (blockDim.x >> 5) + wave;
  if (job >= totalJobs) return;                     // wave-uniform
  const int tm = job / nGroups;
  const int cg = job - tm * nGroups;
  const int row0 = tm << 4;
  const int col0 = cg << 6;

  const int half = lane >> 4;   // K-half selector
  const int mr   = lane & 15;   // A: row-in-tile; B/D: col-in-tile

  const __bf16* aRow = Ab + (long long)(row0 + mr) * K;
  const __bf16* xRow = Xb + (long long)(row0 + mr) * K;
  const __bf16* wl0  = Wlt + (long long)(col0 + mr) * K;
  const __bf16* wr0  = Wrt + (long long)(col0 + mr) * K;

  v8f acc[4] = {};
  for (int kk = 0; kk < K; kk += 32) {
    // A fragment (16x32): elements 0-15 -> K = kk + 16*half + 0..15
    const int ka = kk + half * 16;
    v8bf aMl = *(const v8bf*)(aRow + ka);
    v8bf aMh = *(const v8bf*)(aRow + ka + 8);
    v8bf aXl = *(const v8bf*)(xRow + ka);
    v8bf aXh = *(const v8bf*)(xRow + ka + 8);
    v16bf aM = __builtin_shufflevector(aMl, aMh, 0,1,2,3,4,5,6,7,8,9,10,11,12,13,14,15);
    v16bf aX = __builtin_shufflevector(aXl, aXh, 0,1,2,3,4,5,6,7,8,9,10,11,12,13,14,15);

    // B fragment (32x16): lane col n, elements 0-15 -> K = kk + 16*half + 0..15
    const int kb = kk + half * 16;
#pragma unroll
    for (int nt = 0; nt < 4; ++nt) {
      const long long off = (long long)(nt * 16) * K + kb;
      v16bf bL = *(const v16bf*)(wl0 + off);
      v16bf bR = *(const v16bf*)(wr0 + off);
      acc[nt] = __builtin_amdgcn_wmma_f32_16x16x32_bf16(false, aM, false, bL,
                                                        (short)0, acc[nt], false, false);
      acc[nt] = __builtin_amdgcn_wmma_f32_16x16x32_bf16(false, aX, false, bR,
                                                        (short)0, acc[nt], false, false);
    }
  }

  // D layout: vgpr r -> row = row0 + 8*half + r, col = col0 + nt*16 + mr
#pragma unroll
  for (int nt = 0; nt < 4; ++nt) {
    const int ocol = col0 + nt * 16 + mr;
    const float b = bias[ocol];
#pragma unroll
    for (int r = 0; r < 8; ++r) {
      float v = acc[nt][r] + b;
      if (relu) v = fmaxf(v, 0.0f);
      const long long oi = (long long)(row0 + half * 8 + r) * N + ocol;
      if (outB) outB[oi] = cvt_bf16(v);
      else      outF[oi] = v;
    }
  }
}

// ---------------------------------------------------------------------------
extern "C" void kernel_launch(void* const* d_in, const int* in_sizes, int n_in,
                              void* d_out, int out_size, void* d_ws, size_t ws_size,
                              hipStream_t stream) {
  const float* x   = (const float*)d_in[0];
  const int*   ei  = (const int*)d_in[1];
  const float* Wl1 = (const float*)d_in[2];
  const float* Wr1 = (const float*)d_in[3];
  const float* b1  = (const float*)d_in[4];
  const float* Wl2 = (const float*)d_in[5];
  const float* Wr2 = (const float*)d_in[6];
  const float* b2  = (const float*)d_in[7];
  const int* src = ei;
  const int* dst = ei + N_EDGES;
  float* out = (float*)d_out;
  (void)ws_size; (void)n_in; (void)in_sizes; (void)out_size;

  // ---- workspace arena (all offsets 256B-aligned) ----
  char* base = (char*)d_ws;
  float*  cnt  = (float*)base;                                    base += 50048 * 4;
  float*  agg  = (float*)base;                                    base += (long long)N_NODES * HID_CH * 4;
  __bf16* ab   = (__bf16*)base;                                   base += (long long)N_NODES * HID_CH * 2;
  __bf16* sb   = (__bf16*)base;                                   base += (long long)N_NODES * HID_CH * 2; // xb (L1)
  __bf16* Wl1t = (__bf16*)base;                                   base += IN_CH  * HID_CH * 2;
  __bf16* Wr1t = (__bf16*)base;                                   base += IN_CH  * HID_CH * 2;
  __bf16* Wl2t = (__bf16*)base;                                   base += HID_CH * OUT_CH * 2;
  __bf16* Wr2t = (__bf16*)base;                                   base += HID_CH * OUT_CH * 2;

  const int BLK = 256;

  // ---- weights -> bf16 transposed [N,K] ----
  w_to_bf16t<<<(IN_CH * HID_CH + BLK - 1) / BLK, BLK, 0, stream>>>(Wl1, Wl1t, IN_CH, HID_CH);
  w_to_bf16t<<<(IN_CH * HID_CH + BLK - 1) / BLK, BLK, 0, stream>>>(Wr1, Wr1t, IN_CH, HID_CH);
  w_to_bf16t<<<(HID_CH * OUT_CH + BLK - 1) / BLK, BLK, 0, stream>>>(Wl2, Wl2t, HID_CH, OUT_CH);
  w_to_bf16t<<<(HID_CH * OUT_CH + BLK - 1) / BLK, BLK, 0, stream>>>(Wr2, Wr2t, HID_CH, OUT_CH);

  // ---- layer 1 ----
  zero_f32<<<4096, BLK, 0, stream>>>(cnt, 50048 + (long long)N_NODES * IN_CH);
  degree_kernel<<<(N_EDGES + BLK - 1) / BLK, BLK, 0, stream>>>(dst, cnt, N_EDGES);
  {
    long long tot = (long long)N_EDGES * (IN_CH / 4);
    scatter_add_f32<<<(int)((tot + BLK - 1) / BLK), BLK, 0, stream>>>(x, src, dst, agg,
                                                                     N_EDGES, IN_CH);
  }
  {
    long long ne = (long long)N_NODES * IN_CH;                       // x -> bf16
    rows_to_bf16<<<(int)((ne / 4 + BLK - 1) / BLK), BLK, 0, stream>>>(x, nullptr, sb, ne, IN_CH);
    rows_to_bf16<<<(int)((ne / 4 + BLK - 1) / BLK), BLK, 0, stream>>>(agg, cnt, ab, ne, IN_CH);
  }
  {   // hb = relu(mean@Wl1 + x@Wr1 + b1), bf16 output into its own slot
    int jobs = (N_NODES / 16) * (HID_CH / 64);                        // 12500 waves
    // output hb overwrites... must NOT overwrite sb (xb) while reading it; write to a
    // dedicated region: reuse ab? ab is also an input. Use agg region re-cast after inputs
    // are fully consumed is unsafe under async execution -> give hb its own slot:
    __bf16* hb = (__bf16*)base;  // extra 25.6MB slot
    sage_gemm_bf16<<<(jobs + 7) / 8, BLK, 0, stream>>>(ab, sb, Wl1t, Wr1t, b1,
                                                       nullptr, hb,
                                                       N_NODES, IN_CH, HID_CH, 1);
    // ---- layer 2 ----
    zero_f32<<<4096, BLK, 0, stream>>>(agg, (long long)N_NODES * HID_CH);
    {
      long long tot = (long long)N_EDGES * (HID_CH / 4);
      scatter_add_bf16<<<(int)((tot + BLK - 1) / BLK), BLK, 0, stream>>>(hb, src, dst, agg,
                                                                        N_EDGES, HID_CH);
    }
    {
      long long ne = (long long)N_NODES * HID_CH;
      rows_to_bf16<<<(int)((ne / 4 + BLK - 1) / BLK), BLK, 0, stream>>>(agg, cnt, ab, ne, HID_CH);
    }
    {
      int jobs2 = (N_NODES / 16) * (OUT_CH / 64);                     // 6250 waves
      sage_gemm_bf16<<<(jobs2 + 7) / 8, BLK, 0, stream>>>(ab, hb, Wl2t, Wr2t, b2,
                                                          out, nullptr,
                                                          N_NODES, HID_CH, OUT_CH, 0);
    }
  }
}
